// SwinCrossAttention_65146063946568
// MI455X (gfx1250) — matrix operand order repaired and checked by reading, not compiled, not run on hardware
//
#include <hip/hip_runtime.h>
#include <hip/hip_bf16.h>
#include <cstdint>
#include <cstddef>

// ---------------------------------------------------------------------------
// Swin cross attention on gfx1250 (MI455X), wave32 + v_wmma_f32_16x16x32_f16.
// Shapes: B=16, H=W=64, C=512, HEADS=16, HEAD_DIM=32, WS=8, N=64 tokens/window.
// ---------------------------------------------------------------------------

typedef _Float16 f16;
typedef __attribute__((ext_vector_type(16))) _Float16 v16h;
typedef __attribute__((ext_vector_type(8)))  float    v8f;

union F16Frag { v16h h; unsigned u[8]; };
union U4H8   { uint4 v; f16 h[8]; };

static __device__ __forceinline__ v8f wmma16(const F16Frag& a, const F16Frag& b, v8f c) {
    // D = A(16x32 f16) * B(32x16 f16) + C(16x16 f32)
    return __builtin_amdgcn_wmma_f32_16x16x32_f16(false, a.h, false, b.h,
                                                  (short)0, c, false, false);
}

#define DIMC   512
#define MROWS  65536        // 16 * 4096 tokens
#define SCALE_Q 0.1767766952966369f   // 32^-0.5

// ---------------------------------------------------------------------------
// LayerNorm (f32 in -> f16 out).  One wave32 per 512-elem row, float4 loads.
// ---------------------------------------------------------------------------
__global__ __launch_bounds__(256)
void k_layernorm(const float* __restrict__ x, const float* __restrict__ w,
                 const float* __restrict__ b, f16* __restrict__ out) {
    int row  = blockIdx.x * 8 + (threadIdx.x >> 5);
    int lane = threadIdx.x & 31;
    const float4* xr = (const float4*)(x + (size_t)row * DIMC);

    float4 v[4];
    float s = 0.f;
#pragma unroll
    for (int i = 0; i < 4; ++i) {
        v[i] = xr[lane + 32 * i];
        s += v[i].x + v[i].y + v[i].z + v[i].w;
    }
#pragma unroll
    for (int o = 16; o >= 1; o >>= 1) s += __shfl_xor(s, o, 32);
    float mu = s * (1.0f / DIMC);

    float vs = 0.f;
#pragma unroll
    for (int i = 0; i < 4; ++i) {
        float dx = v[i].x - mu, dy = v[i].y - mu, dz = v[i].z - mu, dw = v[i].w - mu;
        vs += dx * dx + dy * dy + dz * dz + dw * dw;
    }
#pragma unroll
    for (int o = 16; o >= 1; o >>= 1) vs += __shfl_xor(vs, o, 32);
    float rs = rsqrtf(vs * (1.0f / DIMC) + 1e-5f);

    f16* orow = out + (size_t)row * DIMC;
#pragma unroll
    for (int i = 0; i < 4; ++i) {
        int c = (lane + 32 * i) * 4;
        const float* pv = (const float*)&v[i];
#pragma unroll
        for (int j = 0; j < 4; ++j)
            orow[c + j] = (f16)((pv[j] - mu) * rs * w[c + j] + b[c + j]);
    }
}

// ---------------------------------------------------------------------------
// Weight slice/convert f32 -> f16 (optionally scaled).  ncols = 1<<nshift.
// ---------------------------------------------------------------------------
__global__ __launch_bounds__(256)
void k_cvt_w(const float* __restrict__ in, f16* __restrict__ out,
             int in_stride, int col_off, int nshift, float scale, int total) {
    int i = blockIdx.x * 256 + threadIdx.x;
    if (i >= total) return;
    int k = i >> nshift, n = i & ((1 << nshift) - 1);
    out[i] = (f16)(in[(size_t)k * in_stride + col_off + n] * scale);
}

__global__ __launch_bounds__(256)
void k_cvt_b(const float* __restrict__ in, float* __restrict__ out,
             int off, int n, float scale) {
    int i = blockIdx.x * 256 + threadIdx.x;
    if (i < n) out[i] = in[off + i] * scale;
}

// Relative-position bias table: bt[h][qi][kj] = rpb[((dy+7)*15 + (dx+7))*16 + h]
__global__ __launch_bounds__(256)
void k_biastab(const float* __restrict__ rpb, float* __restrict__ bt) {
    int i = blockIdx.x * 256 + threadIdx.x;          // 16*64*64 = 65536
    int h = i >> 12, rest = i & 4095;
    int qi = rest >> 6, kj = rest & 63;
    int dy = (qi >> 3) - (kj >> 3);
    int dx = (qi & 7) - (kj & 7);
    int idx = (dy + 7) * 15 + (dx + 7);
    bt[i] = rpb[idx * 16 + h];
}

// ---------------------------------------------------------------------------
// WMMA GEMM: C[M x N] = A[M x 512](f16) * W[512 x N](f16) + bias.
// Block tile 128x64, 8 waves, each wave 32x32 (2x2 WMMA tiles), K step 32.
// RESID=false: f16 store.  RESID=true: f32 store with shortcut add.
// ---------------------------------------------------------------------------
template <bool RESID>
__global__ __launch_bounds__(256)
void k_gemm(const f16* __restrict__ A, const f16* __restrict__ W,
            const float* __restrict__ bias, int N,
            f16* __restrict__ outh, float* __restrict__ outf,
            const float* __restrict__ shortcut) {
    __shared__ __align__(16) f16 As[128][32];   // [m][k] row-major
    __shared__ __align__(16) f16 Bt[64][32];    // [n][k] (transposed)
    const unsigned* Asu = (const unsigned*)&As[0][0];  // row stride 16 uints
    const unsigned* Btu = (const unsigned*)&Bt[0][0];  // row stride 16 uints

    int tid  = threadIdx.x;
    int lane = tid & 31, wv = tid >> 5;
    int mw = wv >> 1, nw = wv & 1;              // 4x2 wave grid
    int m0 = blockIdx.y * 128, n0 = blockIdx.x * 64;

    // A fill: thread -> row tid/2, 16 halves at col (tid&1)*16
    int ar = tid >> 1, ac = (tid & 1) << 4;
    // B fill: thread -> k = tid/8, 8 halves at n (tid&7)*8
    int bk = tid >> 3, bn = (tid & 7) << 3;

    v8f acc[2][2];
#pragma unroll
    for (int mt = 0; mt < 2; ++mt)
#pragma unroll
        for (int nt = 0; nt < 2; ++nt) acc[mt][nt] = (v8f){};

    for (int kk = 0; kk < DIMC; kk += 32) {
        __syncthreads();
        {   // 128x32 A tile: 2 x b128 global loads + 2 x b128 LDS stores
            const uint4* src = (const uint4*)(A + (size_t)(m0 + ar) * DIMC + kk + ac);
            uint4 q0 = src[0], q1 = src[1];
            *(uint4*)&As[ar][ac]     = q0;
            *(uint4*)&As[ar][ac + 8] = q1;
        }
        {   // 32x64 W tile, transposed into Bt: b128 load + 8 scalar ds stores
            U4H8 t;
            t.v = *(const uint4*)(W + (size_t)(kk + bk) * N + n0 + bn);
#pragma unroll
            for (int j = 0; j < 8; ++j) Bt[bn + j][bk] = t.h[j];
        }
        __syncthreads();

        F16Frag a[2], b[2];
#pragma unroll
        for (int mt = 0; mt < 2; ++mt) {
            int m = mw * 32 + mt * 16 + (lane & 15);
            int base = m * 16 + ((lane >> 4) << 3);       // kbase/2 = 8 for hi half
#pragma unroll
            for (int j = 0; j < 8; ++j) a[mt].u[j] = Asu[base + j];
        }
#pragma unroll
        for (int nt = 0; nt < 2; ++nt) {
            int n = nw * 32 + nt * 16 + (lane & 15);
            int base = n * 16 + ((lane >> 4) << 3);       // kb/2 = 8 for hi half
#pragma unroll
            for (int j = 0; j < 8; ++j) b[nt].u[j] = Btu[base + j];
        }
#pragma unroll
        for (int mt = 0; mt < 2; ++mt)
#pragma unroll
            for (int nt = 0; nt < 2; ++nt)
                acc[mt][nt] = wmma16(a[mt], b[nt], acc[mt][nt]);
    }

    // epilogue: C layout — lane n = lane&15 (+16 -> M+8), vgpr r = M row within 8
#pragma unroll
    for (int mt = 0; mt < 2; ++mt)
#pragma unroll
        for (int nt = 0; nt < 2; ++nt) {
            int mb = m0 + mw * 32 + mt * 16 + ((lane >> 4) << 3);
            int n  = n0 + nw * 32 + nt * 16 + (lane & 15);
            float bn_ = bias[n];
#pragma unroll
            for (int r = 0; r < 8; ++r) {
                size_t idx = (size_t)(mb + r) * N + n;
                float val = acc[mt][nt][r] + bn_;
                if (RESID) outf[idx] = val + shortcut[idx];
                else       outh[idx] = (f16)val;
            }
        }
}

// ---------------------------------------------------------------------------
// Attention: one block per (window, head).  4 waves / 128 threads.
//  S = Q Kt + bias  -> softmax rows -> O = P V  (all via WMMA f16)
// ---------------------------------------------------------------------------
__global__ __launch_bounds__(128)
void k_attn(const f16* __restrict__ Q, const f16* __restrict__ KV,
            const float* __restrict__ biasTab, f16* __restrict__ aout) {
    __shared__ __align__(16) f16  Qs[64][32];   // [query][d]
    __shared__ __align__(16) f16  Ks[64][32];   // [key][d]   (== Kt fragment layout)
    __shared__ __align__(16) f16  Vt[32][64];   // [d][key]
    __shared__           float    Ss[64][64];
    __shared__ __align__(16) f16  Ps[64][64];

    int wi = blockIdx.x, h = blockIdx.y;
    int tid = threadIdx.x, lane = tid & 31, wv = tid >> 5;
    int b = wi >> 6, w = wi & 63, wy = w >> 3, wx = w & 7;

    // gather window tokens from raster-ordered Q / KV (vectorized b128)
    {
        int t = tid >> 1, c = (tid & 1) << 4;            // token, half-chunk
        int ty = t >> 3, tx = t & 7;
        size_t grow = (size_t)b * 4096 + (size_t)((wy * 8 + ty) * 64 + wx * 8 + tx);
        const uint4* qsrc = (const uint4*)(Q  + grow * 512  + h * 32 + c);
        const uint4* ksrc = (const uint4*)(KV + grow * 1024 + h * 32 + c);
        *(uint4*)&Qs[t][c]     = qsrc[0];
        *(uint4*)&Qs[t][c + 8] = qsrc[1];
        *(uint4*)&Ks[t][c]     = ksrc[0];
        *(uint4*)&Ks[t][c + 8] = ksrc[1];
        U4H8 v0, v1;
        const uint4* vsrc = (const uint4*)(KV + grow * 1024 + 512 + h * 32 + c);
        v0.v = vsrc[0]; v1.v = vsrc[1];
#pragma unroll
        for (int j = 0; j < 8; ++j) { Vt[c + j][t] = v0.h[j]; Vt[c + 8 + j][t] = v1.h[j]; }
    }
    __syncthreads();

    const unsigned* Qu = (const unsigned*)&Qs[0][0];   // stride 16 uints
    const unsigned* Ku = (const unsigned*)&Ks[0][0];   // stride 16 uints

    // S = Q * K^T  (wave wv owns query-tile wv; loops over 4 key-tiles)
    F16Frag qa;
    {
        int m = wv * 16 + (lane & 15);
        int base = m * 16 + ((lane >> 4) << 3);
#pragma unroll
        for (int j = 0; j < 8; ++j) qa.u[j] = Qu[base + j];
    }
#pragma unroll
    for (int nb = 0; nb < 4; ++nb) {
        F16Frag kb;
        int n = nb * 16 + (lane & 15);
        int base = n * 16 + ((lane >> 4) << 3);
#pragma unroll
        for (int j = 0; j < 8; ++j) kb.u[j] = Ku[base + j];
        v8f acc = (v8f){};
        acc = wmma16(qa, kb, acc);
        int mm0 = wv * 16 + ((lane >> 4) << 3);
        int nn  = nb * 16 + (lane & 15);
#pragma unroll
        for (int r = 0; r < 8; ++r)
            Ss[mm0 + r][nn] = acc[r] + biasTab[(h * 64 + mm0 + r) * 64 + nn];
    }
    __syncthreads();

    // row softmax (64 rows, threads 0..63)
    if (tid < 64) {
        float mx = -1e30f;
#pragma unroll 4
        for (int j = 0; j < 64; ++j) mx = fmaxf(mx, Ss[tid][j]);
        float sum = 0.f;
#pragma unroll 4
        for (int j = 0; j < 64; ++j) { float e = __expf(Ss[tid][j] - mx); Ss[tid][j] = e; sum += e; }
        float inv = 1.f / sum;
#pragma unroll 4
        for (int j = 0; j < 64; ++j) Ps[tid][j] = (f16)(Ss[tid][j] * inv);
    }
    __syncthreads();

    const unsigned* Pu = (const unsigned*)&Ps[0][0];   // stride 32 uints
    const unsigned* Vu = (const unsigned*)&Vt[0][0];   // stride 32 uints

    // O = P * V  : output tile 16x16, K=64 in two 32-chunks
#pragma unroll
    for (int dt = 0; dt < 2; ++dt) {
        v8f acc = (v8f){};
#pragma unroll
        for (int kc = 0; kc < 2; ++kc) {
            F16Frag pa, vb;
            int m = wv * 16 + (lane & 15);
            int pbase = m * 32 + kc * 16 + ((lane >> 4) << 3);
#pragma unroll
            for (int j = 0; j < 8; ++j) pa.u[j] = Pu[pbase + j];
            int d = dt * 16 + (lane & 15);
            int vbase = d * 32 + kc * 16 + ((lane >> 4) << 3);
#pragma unroll
            for (int j = 0; j < 8; ++j) vb.u[j] = Vu[vbase + j];
            acc = wmma16(pa, vb, acc);
        }
#pragma unroll
        for (int r = 0; r < 8; ++r) {
            int mm = wv * 16 + r + ((lane >> 4) << 3);
            int dd = dt * 16 + (lane & 15);
            int ty = mm >> 3, tx = mm & 7;
            size_t grow = (size_t)b * 4096 + (size_t)((wy * 8 + ty) * 64 + wx * 8 + tx);
            aout[grow * 512 + h * 32 + dd] = (f16)acc[r];
        }
    }
}

// ---------------------------------------------------------------------------
// Launch.  Workspace layout (bytes, requires ~323 MiB):
//   [0,64Mi)    x1n f16  (reused as attention-out after KV GEMM)
//   [64,128Mi)  x2n f16
//   [128,192Mi) Q f16
//   [192,320Mi) KV f16 (K cols 0..511, V cols 512..1023)
//   [320Mi..)   Wq/Wkv/Wp f16, bq/bkv/bp f32, bias table f32
// ---------------------------------------------------------------------------
extern "C" void kernel_launch(void* const* d_in, const int* in_sizes, int n_in,
                              void* d_out, int out_size, void* d_ws, size_t ws_size,
                              hipStream_t stream) {
    const float* x1    = (const float*)d_in[0];
    const float* x2    = (const float*)d_in[1];
    const float* n1w   = (const float*)d_in[2];
    const float* n1b   = (const float*)d_in[3];
    const float* n2w   = (const float*)d_in[4];
    const float* n2b   = (const float*)d_in[5];
    const float* qkv1w = (const float*)d_in[6];
    const float* qkv1b = (const float*)d_in[7];
    const float* qkv2w = (const float*)d_in[8];
    const float* qkv2b = (const float*)d_in[9];
    const float* rpb   = (const float*)d_in[10];
    const float* projw = (const float*)d_in[11];
    const float* projb = (const float*)d_in[12];

    char* ws = (char*)d_ws;
    f16* x1n = (f16*)(ws);
    f16* x2n = (f16*)(ws + (64ull  << 20));
    f16* Qb  = (f16*)(ws + (128ull << 20));
    f16* KVb = (f16*)(ws + (192ull << 20));
    char* sm = ws + (320ull << 20);
    f16*   Wq   = (f16*)(sm);                          // 512*512  f16
    f16*   Wkv  = (f16*)(sm + (512ull  << 10));        // 512*1024 f16
    f16*   Wp   = (f16*)(sm + (1536ull << 10));        // 512*512  f16
    float* bq   = (float*)(sm + (2048ull << 10));      // 512 f32
    float* bkv  = bq + 512;                            // 1024 f32
    float* bp   = bkv + 1024;                          // 512 f32
    float* btab = bp + 512;                            // 16*64*64 f32
    f16*   aout = x1n;                                 // reuse after KV GEMM

    // 1) LayerNorm -> f16
    k_layernorm<<<MROWS / 8, 256, 0, stream>>>(x1, n1w, n1b, x1n);
    k_layernorm<<<MROWS / 8, 256, 0, stream>>>(x2, n2w, n2b, x2n);

    // 2) weight / bias prep
    k_cvt_w<<<(512 * 512)  / 256, 256, 0, stream>>>(qkv2w, Wq,  1536, 0,   9,  SCALE_Q, 512 * 512);
    k_cvt_w<<<(512 * 1024) / 256, 256, 0, stream>>>(qkv1w, Wkv, 1536, 512, 10, 1.0f,    512 * 1024);
    k_cvt_w<<<(512 * 512)  / 256, 256, 0, stream>>>(projw, Wp,  512,  0,   9,  1.0f,    512 * 512);
    k_cvt_b<<<2, 256, 0, stream>>>(qkv2b, bq,  0,   512,  SCALE_Q);
    k_cvt_b<<<4, 256, 0, stream>>>(qkv1b, bkv, 512, 1024, 1.0f);
    k_cvt_b<<<2, 256, 0, stream>>>(projb, bp,  0,   512,  1.0f);
    k_biastab<<<65536 / 256, 256, 0, stream>>>(rpb, btab);

    // 3) QKV projections (WMMA)
    k_gemm<false><<<dim3(8,  MROWS / 128), 256, 0, stream>>>(x2n, Wq,  bq,  512,  Qb,  nullptr, nullptr);
    k_gemm<false><<<dim3(16, MROWS / 128), 256, 0, stream>>>(x1n, Wkv, bkv, 1024, KVb, nullptr, nullptr);

    // 4) windowed cross-attention (WMMA)
    k_attn<<<dim3(1024, 16), 128, 0, stream>>>(Qb, KVb, btab, aout);

    // 5) output projection + residual (WMMA, f32 epilogue)
    k_gemm<true><<<dim3(8, MROWS / 128), 256, 0, stream>>>(aout, Wp, bp, 512,
                                                           nullptr, (float*)d_out, x1);
}
